// LinearQKVAttention_46703474377050
// MI455X (gfx1250) — matrix operand order repaired and checked by reading, not compiled, not run on hardware
//
#include <hip/hip_runtime.h>

typedef __attribute__((ext_vector_type(16))) _Float16 v16h;
typedef __attribute__((ext_vector_type(8)))  _Float16 v8h;
typedef __attribute__((ext_vector_type(2)))  _Float16 h2;
typedef __attribute__((ext_vector_type(8)))  float    v8f;

static constexpr int kBH  = 128;   // B*H
static constexpr int kN   = 4096;  // sequence
static constexpr int kD   = 64;    // feature (== E)
static constexpr float kEps = 1e-6f;

// Load 16 contiguous f16 from LDS as two 16B-aligned b128 loads.
__device__ __forceinline__ v16h ld16(const _Float16* p) {
  v8h lo = *(const v8h*)p;
  v8h hi = *(const v8h*)(p + 8);
  return __builtin_shufflevector(lo, hi, 0, 1, 2, 3, 4, 5, 6, 7,
                                 8, 9, 10, 11, 12, 13, 14, 15);
}

// WMMA A fragment (16x32 f16): row = M, elements step K.
// lane<16: K {kc..kc+15}; lane>=16: K {kc+16..kc+31}.
__device__ __forceinline__ v16h ldA(const _Float16* row, int kc, int lh) {
  v8h lo = *(const v8h*)(row + kc + lh * 16);
  v8h hi = *(const v8h*)(row + kc + 8 + lh * 16);
  return __builtin_shufflevector(lo, hi, 0, 1, 2, 3, 4, 5, 6, 7,
                                 8, 9, 10, 11, 12, 13, 14, 15);
}

// ---------------------------------------------------------------------------
// Kernel 1: ctx[bh][d][e] += sum_n exp(k[n][d]) * v[n][e]
//           colsum[bh][d] += sum_n exp(k[n][d])
// grid = 128 bh * 8 splits of 512 rows; 256 threads (8 waves).
// Both operands stored K(n)-contiguous in LDS for b128 fragment loads.
// ---------------------------------------------------------------------------
__global__ __launch_bounds__(256) void k_ctx(const float* __restrict__ Kp,
                                             const float* __restrict__ Vp,
                                             float* __restrict__ ctx,
                                             float* __restrict__ colsum) {
  __shared__ _Float16 sEt[64][136];  // exp(k)^T: [d][n_local]  (pitch 272B)
  __shared__ _Float16 sVt[64][136];  // v^T:      [e][n_local]
  __shared__ float    sSum[64];

  const int t     = threadIdx.x;
  const int bh    = blockIdx.x >> 3;
  const int split = blockIdx.x & 7;
  const size_t base = ((size_t)bh * kN + (size_t)split * 512) * kD;

  const int lane = t & 31;
  const int wave = t >> 5;
  const int lh   = lane >> 4;
  const int l16  = lane & 15;
  const int dt   = wave & 3;         // d-tile (0..3)
  const int et0  = (wave >> 2) * 2;  // e-tiles et0, et0+1
  const int dfix = t & 63;

  if (t < 64) sSum[t] = 0.0f;

  v8f acc0 = {};
  v8f acc1 = {};
  float csum = 0.0f;

  const _Float16* arow  = &sEt[dt * 16 + l16][0];
  const _Float16* brow0 = &sVt[et0 * 16 + l16][0];
  const _Float16* brow1 = &sVt[(et0 + 1) * 16 + l16][0];

  for (int c = 0; c < 4; ++c) {            // 4 sub-chunks of 128 rows
    const size_t cbase = base + (size_t)c * 128 * kD;
    __syncthreads();                       // LDS reuse fence (covers sSum init)

    // exp(k) -> sEt transposed; rows processed in (n, n+1) pairs -> b32 stores
    #pragma unroll
    for (int i = 0; i < 16; ++i) {
      const int nl = 2 * (t >> 6) + i * 8;         // even, pairs cover 0..127
      const size_t g = cbase + (size_t)nl * kD + dfix;
      const float e0 = __expf(Kp[g]);
      const float e1 = __expf(Kp[g + kD]);
      csum += e0 + e1;
      h2 p; p.x = (_Float16)e0; p.y = (_Float16)e1;
      *(h2*)&sEt[dfix][nl] = p;
    }
    // v -> sVt transposed (same pairing)
    #pragma unroll
    for (int i = 0; i < 16; ++i) {
      const int nl = 2 * (t >> 6) + i * 8;
      const size_t g = cbase + (size_t)nl * kD + dfix;
      h2 p; p.x = (_Float16)Vp[g]; p.y = (_Float16)Vp[g + kD];
      *(h2*)&sVt[dfix][nl] = p;
    }
    __syncthreads();

    #pragma unroll
    for (int kc = 0; kc < 128; kc += 32) {
      const v16h a  = ldA(arow, kc, lh);
      const v16h b0 = ld16(brow0 + kc + lh * 16);
      const v16h b1 = ld16(brow1 + kc + lh * 16);
      acc0 = __builtin_amdgcn_wmma_f32_16x16x32_f16(false, a, false, b0,
                                                    (short)0, acc0, false, false);
      acc1 = __builtin_amdgcn_wmma_f32_16x16x32_f16(false, a, false, b1,
                                                    (short)0, acc1, false, false);
    }
  }

  // column sums: LDS combine, then one global atomic per d
  atomicAdd(&sSum[dfix], csum);
  __syncthreads();
  if (t < 64) atomicAdd(&colsum[bh * 64 + t], sSum[t]);

  // partial 64x64 ctx -> global f32 atomics (C/D layout: VGPR r = row r+8*lh)
  const size_t cb = (size_t)bh * 4096;
  #pragma unroll
  for (int r = 0; r < 8; ++r) {
    const int dr = dt * 16 + r + lh * 8;
    atomicAdd(&ctx[cb + dr * 64 + et0 * 16 + l16],       acc0[r]);
    atomicAdd(&ctx[cb + dr * 64 + (et0 + 1) * 16 + l16], acc1[r]);
  }
}

// ---------------------------------------------------------------------------
// Kernel 2: ctxnT[bh][e][d] = ctx[d][e] / (colsum[d] + eps)   (f16, transposed)
// ---------------------------------------------------------------------------
__global__ __launch_bounds__(256) void k_norm(const float* __restrict__ ctx,
                                              const float* __restrict__ colsum,
                                              _Float16* __restrict__ ctxnT) {
  __shared__ float sT[64][65];   // transpose staging, conflict-free pitch
  __shared__ float sInv[64];

  const int bh = blockIdx.x;
  const int t  = threadIdx.x;
  const size_t cb = (size_t)bh * 4096;

  if (t < 64) sInv[t] = 1.0f / (colsum[bh * 64 + t] + kEps);
  #pragma unroll
  for (int i = 0; i < 16; ++i) {        // coalesced read, transposed LDS write
    const int idx = t + i * 256;
    sT[idx & 63][idx >> 6] = ctx[cb + idx];   // sT[e][d]
  }
  __syncthreads();
  #pragma unroll
  for (int i = 0; i < 8; ++i) {         // packed b32 coalesced f16 writes
    const int idx = 2 * t + i * 512;
    const int e = idx >> 6, d = idx & 63;
    h2 p;
    p.x = (_Float16)(sT[e][d] * sInv[d]);
    p.y = (_Float16)(sT[e][d + 1] * sInv[d + 1]);
    *(h2*)&ctxnT[cb + idx] = p;
  }
}

// ---------------------------------------------------------------------------
// Kernel 3: out[n][e] = softmax_row(q)[n][:] @ ctxn   (M=128/block, N=64, K=64)
// grid = 128 bh * 32 chunks of 128 rows; 256 threads (8 waves).
// ---------------------------------------------------------------------------
__global__ __launch_bounds__(256) void k_out(const float* __restrict__ Qp,
                                             const _Float16* __restrict__ ctxnT,
                                             float* __restrict__ out) {
  __shared__ _Float16 sCt[64][72];    // context^T: [e][d] (pitch 144B)
  __shared__ float    sQf[128][64];   // exp(q)
  __shared__ _Float16 sQh[128][72];   // softmaxed q (WMMA A), pitch 144B
  __shared__ float    sRow[128];

  const int t   = threadIdx.x;
  const int bh  = blockIdx.x >> 5;
  const int n0  = (blockIdx.x & 31) * 128;
  const size_t qbase = ((size_t)bh * kN + n0) * kD;
  const size_t cb    = (size_t)bh * 4096;

  if (t < 128) sRow[t] = 0.0f;

  // context tile: packed b32 global reads -> contiguous LDS rows
  #pragma unroll
  for (int i = 0; i < 8; ++i) {
    const int idx = 2 * t + i * 512;
    *(h2*)&sCt[idx >> 6][idx & 63] = *(const h2*)&ctxnT[cb + idx];
  }

  // exp(q): float2 coalesced loads, b64 LDS stores
  #pragma unroll
  for (int i = 0; i < 16; ++i) {
    const int idx = 2 * t + i * 512;
    const int r = idx >> 6, d = idx & 63;
    const float2 qv = *(const float2*)&Qp[qbase + idx];
    sQf[r][d]     = __expf(qv.x);
    sQf[r][d + 1] = __expf(qv.y);
  }
  __syncthreads();

  // row sums: 2 threads/row, one LDS atomic each
  {
    const int r = t >> 1;
    const int h32 = (t & 1) * 32;
    float s = 0.0f;
    #pragma unroll
    for (int j = 0; j < 32; ++j) s += sQf[r][h32 + j];
    atomicAdd(&sRow[r], s);
  }
  __syncthreads();

  // normalize -> f16 A-matrix (packed b32 stores)
  #pragma unroll
  for (int i = 0; i < 16; ++i) {
    const int idx = 2 * t + i * 512;
    const int r = idx >> 6, d = idx & 63;
    const float inv = 1.0f / sRow[r];
    h2 p;
    p.x = (_Float16)(sQf[r][d] * inv);
    p.y = (_Float16)(sQf[r][d + 1] * inv);
    *(h2*)&sQh[r][d] = p;
  }
  __syncthreads();

  // WMMA: each wave owns one 16-row stripe, 4 e-tiles; K=64 in 2 steps
  const int lane = t & 31;
  const int wave = t >> 5;
  const int lh   = lane >> 4;
  const int l16  = lane & 15;
  const _Float16* arow = &sQh[wave * 16 + l16][0];

  #pragma unroll
  for (int et = 0; et < 4; ++et) {
    const _Float16* brow = &sCt[et * 16 + l16][0];
    v8f acc = {};
    #pragma unroll
    for (int ks = 0; ks < 64; ks += 32) {
      const v16h a = ldA(arow, ks, lh);
      const v16h b = ld16(brow + ks + lh * 16);
      acc = __builtin_amdgcn_wmma_f32_16x16x32_f16(false, a, false, b,
                                                   (short)0, acc, false, false);
    }
    #pragma unroll
    for (int r = 0; r < 8; ++r) {
      const int row = n0 + wave * 16 + r + lh * 8;
      out[((size_t)bh * kN + row) * kD + et * 16 + l16] = acc[r];
    }
  }
}

// ---------------------------------------------------------------------------
extern "C" void kernel_launch(void* const* d_in, const int* in_sizes, int n_in,
                              void* d_out, int out_size, void* d_ws, size_t ws_size,
                              hipStream_t stream) {
  (void)in_sizes; (void)n_in; (void)out_size; (void)ws_size;
  const float* q = (const float*)d_in[0];
  const float* k = (const float*)d_in[1];
  const float* v = (const float*)d_in[2];
  float* out = (float*)d_out;

  // ws layout: ctx f32 [128*64*64] | colsum f32 [128*64] | ctxnT f16 [128*64*64]
  float*    ctx    = (float*)d_ws;
  float*    colsum = ctx + kBH * kD * kD;
  _Float16* ctxnT  = (_Float16*)(colsum + kBH * kD);

  // zero the atomic accumulators every launch (graph-capture safe)
  hipMemsetAsync(d_ws, 0, (size_t)(kBH * kD * kD + kBH * kD) * sizeof(float), stream);

  k_ctx <<<dim3(kBH * 8),  dim3(256), 0, stream>>>(k, v, ctx, colsum);
  k_norm<<<dim3(kBH),      dim3(256), 0, stream>>>(ctx, colsum, ctxnT);
  k_out <<<dim3(kBH * 32), dim3(256), 0, stream>>>(q, ctxnT, out);
}
